// SREL_intra_32925219291676
// MI455X (gfx1250) — matrix operand order repaired and checked by reading, not compiled, not run on hardware
//
#include <hip/hip_runtime.h>
#include <hip/hip_bf16.h>
#include <math.h>

#define BB    2048
#define MM    64
#define LSZ   128
#define LWW   128
#define LVV   64
#define HH    256
#define NSTEP 8

typedef __attribute__((ext_vector_type(16))) __bf16 v16bf;
typedef __attribute__((ext_vector_type(8)))  float  v8f;

union Frag16 { v16bf v; int4 q[2]; };
union Pack8  { int4 q; unsigned short u[8]; };

__device__ __forceinline__ unsigned short f2bf(float f) {
  unsigned int u = __float_as_uint(f);
  u += 0x7FFFu + ((u >> 16) & 1u);          // round-to-nearest-even
  return (unsigned short)(u >> 16);
}
__device__ __forceinline__ float bf2f(unsigned short h) {
  return __uint_as_float(((unsigned int)h) << 16);
}

// A fragment (16x32 bf16), A row-major [row][k], ldk in halves (mult of 8).
// lane: row = row0 + (lane&15), khalf = lane>>4.
// elems 0..7  -> K = kbase + khalf*8 + e
// elems 8..15 -> K = kbase + 16 + khalf*8 + (e-8)
__device__ __forceinline__ v16bf load_a_frag(const unsigned short* A, int row0,
                                             int ldk, int kbase, int lane) {
  int r  = row0 + (lane & 15);
  int kh = (lane >> 4) & 1;
  const unsigned short* p = A + (size_t)r * ldk + kbase + kh * 8;
  Frag16 f;
  f.q[0] = *(const int4*)(p);
  f.q[1] = *(const int4*)(p + 16);
  return f.v;
}

// B fragment (32x16 bf16), B stored TRANSPOSED [n][k], ldk in halves (mult of 8).
// lane: col n = n0 + (lane&15); khalf = lane>>4; elems 0..7 -> K = kbase + khalf*8 + e, elems 8..15 -> K = kbase + 16 + khalf*8 + (e-8)
__device__ __forceinline__ v16bf load_b_frag(const unsigned short* Bt, int n0,
                                             int ldk, int kbase, int lane) {
  int n  = n0 + (lane & 15);
  int kh = (lane >> 4) & 1;
  const unsigned short* p = Bt + (size_t)n * ldk + kbase + kh * 8;
  Frag16 f;
  f.q[0] = *(const int4*)(p);
  f.q[1] = *(const int4*)(p + 16);
  return f.v;
}

// ---------------------------------------------------------------------------
// P0: pack/transpose weights to bf16
// ---------------------------------------------------------------------------
__global__ void __launch_bounds__(256) pack_kernel(
    const float* W1e, const float* W1r, const float* W2e,
    unsigned short* W1s_e_t, unsigned short* W1s_r_t,
    unsigned short* W1w_e_t, unsigned short* W1w_r_t,
    unsigned short* W2e_t) {
  int idx = blockIdx.x * 256 + threadIdx.x;
  if (idx < 65536) { int n = idx >> 8, k = idx & 255; W1s_e_t[n*256+k] = f2bf(W1e[k*HH + n]); return; }
  idx -= 65536;
  if (idx < 65536) { int n = idx >> 8, k = idx & 255; W1s_r_t[n*256+k] = f2bf(W1r[k*HH + n]); return; }
  idx -= 65536;
  if (idx < 81920) { int n = idx / 320, k = idx % 320; W1w_e_t[n*320+k] = f2bf(W1e[(256+k)*HH + n]); return; }
  idx -= 81920;
  if (idx < 81920) { int n = idx / 320, k = idx % 320; W1w_r_t[n*320+k] = f2bf(W1r[(256+k)*HH + n]); return; }
  idx -= 81920;
  if (idx < 32768) { int n = idx >> 8, k = idx & 255; W2e_t[n*256+k] = f2bf(W2e[k*LSZ + n]); return; }
}

// ---------------------------------------------------------------------------
// P1: wc[b,m,:] = (wr|wi|vt) @ W1[256:576] + b1  (step-invariant block)
//     [64 x 320] @ [320 x 256] per b, both heads; bf16 WMMA, K=10 steps.
// ---------------------------------------------------------------------------
__global__ void __launch_bounds__(256) wc_kernel(
    const float* wMr, const float* wMi, const float* vM,
    const unsigned short* W1w_e_t, const unsigned short* W1w_r_t,
    const float* b1e, const float* b1r,
    unsigned short* wc_e, unsigned short* wc_r) {
  const int b = blockIdx.x;
  const int t = threadIdx.x;
  const int lane = t & 31, wave = t >> 5;
  __shared__ unsigned short A[64 * 328];   // [m][k], k<320, padded ldk

  for (int i = 0; i < 80; ++i) {           // 64*320 elems, e = k*64+m (coalesced)
    int e = t + i * 256;
    int k = e >> 6, m = e & 63;
    float v;
    if (k < 128)      v = wMr[(size_t)b * (LWW*MM) + k * MM + m];
    else if (k < 256) v = wMi[(size_t)b * (LWW*MM) + (k - 128) * MM + m];
    else              v = vM[(k - 256) * MM + m];
    A[m * 328 + k] = f2bf(v);
  }
  __syncthreads();

  const int mt = wave >> 1;
  const int nb = (wave & 1) * 8;
  v16bf afr[10];
  for (int k8 = 0; k8 < 10; ++k8) afr[k8] = load_a_frag(A, mt * 16, 328, k8 * 32, lane);

  for (int head = 0; head < 2; ++head) {
    const unsigned short* Bt = head ? W1w_r_t : W1w_e_t;
    const float* b1 = head ? b1r : b1e;
    unsigned short* wc = head ? wc_r : wc_e;
    for (int j = 0; j < 8; ++j) {
      int nt = nb + j;
      v8f acc = {};
      for (int k8 = 0; k8 < 10; ++k8) {
        v16bf bf = load_b_frag(Bt, nt * 16, 320, k8 * 32, lane);
        acc = __builtin_amdgcn_wmma_f32_16x16x32_bf16(false, afr[k8], false, bf,
                                                      (short)0, acc, false, false);
      }
      int n  = nt * 16 + (lane & 15);
      int m0 = mt * 16 + ((lane >> 4) & 1) * 8;
      float bias = b1[n];
      for (int i = 0; i < 8; ++i)
        wc[((size_t)b * MM + (m0 + i)) * HH + n] = f2bf(acc[i] + bias);
    }
  }
}

// ---------------------------------------------------------------------------
// Mega kernel: one WG owns batch row b across ALL 8 steps (rows independent).
// Per step: s=exp(i*phi) -> s_list; sc = [cos|sin]@W1s (matvec, both heads);
// H = relu(sc+wc_e) in LDS; rho via VALU dot; etaM = H@W2e^T + b2e via WMMA;
// eta = rho^T etaM via ds_add_f32; phi -= eta (LDS only).
// ---------------------------------------------------------------------------
__global__ void __launch_bounds__(256) mega_kernel(
    const float* phi0,
    const unsigned short* wc_e, const unsigned short* wc_r,
    const unsigned short* W1s_e_t, const unsigned short* W1s_r_t,
    const unsigned short* W2e_t_g,
    const float* b2e, const float* W2r, const float* b2r,
    float* out_slist, float* out_etaM) {
  const int b = blockIdx.x;
  const int t = threadIdx.x;
  const int lane = t & 31, wave = t >> 5;

  extern __shared__ char smem[];
  unsigned short* Bs  = (unsigned short*)smem;   // W2e^T [128][264]
  unsigned short* Hm  = Bs + 128 * 264;          // H_e   [64][264]
  unsigned short* WcE = Hm + 64 * 264;           // wc_e  [64][256] (linear)
  float* fb   = (float*)(WcE + 64 * 256);
  float* S    = fb;          // 256 : [cos(128) | sin(128)]
  float* sce  = fb + 256;    // 256
  float* scr  = fb + 512;    // 256
  float* w2rs = fb + 768;    // 256
  float* part = fb + 1024;   // 256
  float* rho  = fb + 1280;   // 64
  float* eta  = fb + 1344;   // 128
  float* phi  = fb + 1472;   // 128

  // ---- one-time staging ----
  w2rs[t] = W2r[t];
  if (t < 128) phi[t] = phi0[(size_t)b * LSZ + t];
  for (int i = 0; i < 16; ++i) {                 // Bs: 4096 int4
    int c = t + i * 256;
    int4 d = ((const int4*)W2e_t_g)[c];
    int n = c >> 5, kc = c & 31;
    *(int4*)(&Bs[n * 264 + kc * 8]) = d;
  }
  const unsigned short* wce_g = wc_e + (size_t)b * (MM * HH);
  for (int i = 0; i < 8; ++i)                    // WcE: 2048 int4 (kept 8 steps)
    ((int4*)WcE)[t + i * 256] = ((const int4*)wce_g)[t + i * 256];
  const unsigned short* wcr_g = wc_r + (size_t)b * (MM * HH);
  const float bias_r = b2r[0];
  __syncthreads();

  const int mt    = wave & 3;
  const int nbase = (wave >> 2) * 4;

  for (int step = 0; step < NSTEP; ++step) {
    // ---- s = exp(i*phi): emit s_list slot, build S, zero eta ----
    if (t < 128) {
      float p = phi[t];
      float sn, cs;
      __sincosf(p, &sn, &cs);
      size_t o = ((size_t)(b * LSZ + t) * (NSTEP + 1) + step) * 2;
      out_slist[o] = cs; out_slist[o + 1] = sn;
      S[t] = cs; S[128 + t] = sn;
      eta[t] = 0.f;
    }
    __syncthreads();

    // ---- sc matvec: thread t computes output n=t for both heads ----
    {
      const unsigned short* re = W1s_e_t + (size_t)t * 256;
      const unsigned short* rr = W1s_r_t + (size_t)t * 256;
      float se = 0.f, sr = 0.f;
      for (int c = 0; c < 32; ++c) {
        Pack8 de; de.q = ((const int4*)re)[c];
        Pack8 dr; dr.q = ((const int4*)rr)[c];
        for (int u = 0; u < 8; ++u) {
          float sv = S[c * 8 + u];
          se += sv * bf2f(de.u[u]);
          sr += sv * bf2f(dr.u[u]);
        }
      }
      sce[t] = se; scr[t] = sr;
    }
    __syncthreads();

    // ---- H = relu(sce + wc_e) -> LDS bf16 ----
    for (int i = 0; i < 8; ++i) {                // 64 rows * 32 int4
      int c = t + i * 256;
      int m = c >> 5, kc = c & 31;
      Pack8 d; d.q = ((const int4*)WcE)[c];
      Pack8 o;
      for (int j = 0; j < 8; ++j) {
        float v = sce[kc * 8 + j] + bf2f(d.u[j]);
        o.u[j] = f2bf(v > 0.f ? v : 0.f);
      }
      *(int4*)(&Hm[m * 264 + kc * 8]) = o.q;
    }
    // ---- rho partials: 4 threads per antenna row (wc_r from L2) ----
    {
      int m = t >> 2, p = t & 3;
      float s = 0.f;
      for (int j = 0; j < 8; ++j) {
        int kc = p * 8 + j;
        Pack8 d; d.q = ((const int4*)wcr_g)[m * 32 + kc];
        for (int u = 0; u < 8; ++u) {
          int k = kc * 8 + u;
          float v = scr[k] + bf2f(d.u[u]);
          v = v > 0.f ? v : 0.f;
          s += v * w2rs[k];
        }
      }
      part[t] = s;
    }
    __syncthreads();
    if (t < 64)
      rho[t] = part[t*4] + part[t*4+1] + part[t*4+2] + part[t*4+3] + bias_r;
    __syncthreads();

    // ---- WMMA GEMM [64x256]@[256x128]: 4x8 tiles over 8 waves ----
    v16bf afr[8];
    for (int k8 = 0; k8 < 8; ++k8)
      afr[k8] = load_a_frag(Hm, mt * 16, 264, k8 * 32, lane);

    for (int jn = 0; jn < 4; ++jn) {
      int nt = nbase + jn;
      v8f acc = {};
      for (int k8 = 0; k8 < 8; ++k8) {
        v16bf bfr = load_b_frag(Bs, nt * 16, 264, k8 * 32, lane);
        acc = __builtin_amdgcn_wmma_f32_16x16x32_bf16(false, afr[k8], false, bfr,
                                                      (short)0, acc, false, false);
      }
      int l  = nt * 16 + (lane & 15);
      int m0 = mt * 16 + ((lane >> 4) & 1) * 8;
      float bias = b2e[l];
      float psum = 0.f;
      for (int i = 0; i < 8; ++i) {
        float val = acc[i] + bias;
        out_etaM[(((size_t)b * LSZ + l) * MM + (m0 + i)) * NSTEP + step] = val;
        psum += val * rho[m0 + i];
      }
      atomicAdd(&eta[l], psum);                  // ds_add_f32
    }
    __syncthreads();

    // ---- phi update (LDS only) ----
    if (t < 128) phi[t] -= eta[t];
    __syncthreads();
  }

  // ---- final s slot (t = NSTEP) ----
  if (t < 128) {
    float p = phi[t];
    float sn, cs;
    __sincosf(p, &sn, &cs);
    size_t o = ((size_t)(b * LSZ + t) * (NSTEP + 1) + NSTEP) * 2;
    out_slist[o] = cs; out_slist[o + 1] = sn;
  }
}

// ---------------------------------------------------------------------------
extern "C" void kernel_launch(void* const* d_in, const int* in_sizes, int n_in,
                              void* d_out, int out_size, void* d_ws, size_t ws_size,
                              hipStream_t stream) {
  (void)in_sizes; (void)n_in; (void)out_size; (void)ws_size;
  const float* phi0 = (const float*)d_in[0];
  const float* wMr  = (const float*)d_in[1];
  const float* wMi  = (const float*)d_in[2];
  const float* vM   = (const float*)d_in[3];
  const float* W1e  = (const float*)d_in[4];
  const float* b1e  = (const float*)d_in[5];
  const float* W2e  = (const float*)d_in[6];
  const float* b2e  = (const float*)d_in[7];
  const float* W1r  = (const float*)d_in[8];
  const float* b1r  = (const float*)d_in[9];
  const float* W2r  = (const float*)d_in[10];
  const float* b2r  = (const float*)d_in[11];

  char* ws = (char*)d_ws;
  size_t off = 0;
  auto alloc = [&](size_t bytes) -> char* {
    char* p = ws + off;
    off += (bytes + 255) & ~(size_t)255;
    return p;
  };
  unsigned short* wc_e    = (unsigned short*)alloc((size_t)BB * MM * HH * 2);
  unsigned short* wc_r    = (unsigned short*)alloc((size_t)BB * MM * HH * 2);
  unsigned short* W1s_e_t = (unsigned short*)alloc(256 * 256 * 2);
  unsigned short* W1s_r_t = (unsigned short*)alloc(256 * 256 * 2);
  unsigned short* W1w_e_t = (unsigned short*)alloc(256 * 320 * 2);
  unsigned short* W1w_r_t = (unsigned short*)alloc(256 * 320 * 2);
  unsigned short* W2e_t   = (unsigned short*)alloc(128 * 256 * 2);

  float* out_slist = (float*)d_out;                                  // [B,LS,9] cplx
  float* out_etaM  = out_slist + (size_t)BB * LSZ * (NSTEP + 1) * 2; // [B,LS,M,8]

  pack_kernel<<<1280, 256, 0, stream>>>(W1e, W1r, W2e, W1s_e_t, W1s_r_t,
                                        W1w_e_t, W1w_r_t, W2e_t);
  wc_kernel<<<BB, 256, 0, stream>>>(wMr, wMi, vM, W1w_e_t, W1w_r_t, b1e, b1r,
                                    wc_e, wc_r);

  const size_t smem_mega = (size_t)(128 * 264 + 64 * 264 + 64 * 256) * 2 + 1600 * 4;
  mega_kernel<<<BB, 256, smem_mega, stream>>>(phi0, wc_e, wc_r, W1s_e_t, W1s_r_t,
                                              W2e_t, b2e, W2r, b2r,
                                              out_slist, out_etaM);
}
